// NeighborhoodAttention_87454124081806
// MI455X (gfx1250) — matrix operand order repaired and checked by reading, not compiled, not run on hardware
//
#include <hip/hip_runtime.h>
#include <hip/hip_bf16.h>
#include <math.h>

// ---------------------------------------------------------------------------
// CDNA5 (gfx1250) implementation of PETR-style neighborhood attention.
//  - MLPs run on v_wmma_f32_16x16x32_f16 (f16 in, f32 accum)
//  - top-k runs out of an LDS-resident coordinate cache (64KB of 320KB WGP LDS)
//  - softmax/output kernel is pure streaming at the HBM roofline
// ---------------------------------------------------------------------------

typedef __attribute__((ext_vector_type(16))) _Float16 v16h;
typedef __attribute__((ext_vector_type(8)))  float    v8f;

union F16x16 {
    v16h   v;
    _Float16 h[16];
    float4 q[2];     // two 16-byte chunks
};

#define PI2 6.283185307179586f

// ---------------------------------------------------------------------------
// Weight convert + transpose:  W (K x N, f32 row-major) -> Wt (N x K, f16)
// so each WMMA B-fragment lane reads two runs of 8 K-contiguous halves (2 x b128).
// ---------------------------------------------------------------------------
__global__ void convert_wt_kernel(const float* __restrict__ W,
                                  _Float16* __restrict__ Wt,
                                  int K, int N)
{
    int i = blockIdx.x * blockDim.x + threadIdx.x;
    if (i >= K * N) return;
    int k = i / N;
    int n = i - k * N;
    Wt[(size_t)n * K + k] = (_Float16)W[i];
}

// ---------------------------------------------------------------------------
// Fused: posemb2d -> GEMM1(128->256) -> relu -> GEMM2(256->256) (+memory)
// Block: 256 threads (8 waves), tile = 16 rows x 256 cols.
// Each wave owns two 16x16 column tiles.
// Fragment layouts per CDNA5 ISA 7.12.2:
//   A (16x32 f16): lane L, M=L&15; halves 0..7 -> K = kk+(L>>4)*8..+7,
//                  halves 8..15 -> K = kk+16+(L>>4)*8..+7  (2 contiguous b128)
//   B (32x16 f16): lane L, N=L&15; halves 0..7 -> K = kk+(L>>4)*8..+7,
//                  halves 8..15 -> K = kk+16+(L>>4)*8..+7              (2 contiguous b128)
//   C/D (16x16 f32): reg r, lane L -> M=(L>>4)*8+r, N=L&15
// ---------------------------------------------------------------------------
template<bool ADD_MEM>
__global__ __launch_bounds__(256)
void mlp_pe_kernel(const float* __restrict__ coor,      // (Nrow, 2)
                   const _Float16* __restrict__ W1t,    // (256, 128) f16
                   const float* __restrict__ b1,        // (256)
                   const _Float16* __restrict__ W2t,    // (256, 256) f16
                   const float* __restrict__ b2,        // (256)
                   const float* __restrict__ mem,       // (Nrow, 256) or null
                   float* __restrict__ out,             // (Nrow, 256)
                   int Nrow)
{
    __shared__ __align__(16) _Float16 sEmb[16 * 128];   // 4 KB  (A for GEMM1)
    __shared__ __align__(16) _Float16 sHid[16 * 256];   // 8 KB  (A for GEMM2)
    __shared__ __align__(16) float    sOut[16 * 256];   // 16 KB (store staging)

    const int tid  = threadIdx.x;
    const int lane = tid & 31;
    const int wave = tid >> 5;
    const int row0 = blockIdx.x * 16;

    // ---- Stage 1: sinusoidal embedding, 16 rows x 128 feats, write f16 A-tile
    // feature f: f<64 from y, f>=64 from x; within 64: even->sin, odd->cos,
    // freq = 10000^((f>>1)/32)
    for (int v = tid; v < 16 * 128; v += 256) {
        int r   = v >> 7;
        int f   = v & 127;
        int sub = f & 63;
        float c    = coor[(size_t)(row0 + r) * 2 + (f < 64 ? 1 : 0)];
        float freq = __powf(10000.0f, (float)(sub >> 1) * (1.0f / 32.0f));
        float ang  = c * PI2 / freq;
        float e    = ((sub & 1) == 0) ? __sinf(ang) : __cosf(ang);
        sEmb[v] = (_Float16)e;
    }
    __syncthreads();

    const int mrow  = (lane >> 4) << 3;   // 0 or 8 : row base this lane holds in C/D
    const int aoff  = (lane >> 4) << 3;   // A-fragment K sub-offset
    const int boff  = (lane >> 4) << 3;   // B-fragment K sub-offset

    // ---- GEMM1: H = relu(emb @ W1 + b1) -> sHid (f16)
    #pragma unroll
    for (int t = 0; t < 2; ++t) {
        const int ncol = (wave * 2 + t) * 16 + (lane & 15);
        v8f acc = {};
        #pragma unroll
        for (int kk = 0; kk < 128; kk += 32) {
            F16x16 a, b;
            const _Float16* ap = &sEmb[(lane & 15) * 128 + kk + aoff];
            a.q[0] = *(const float4*)(ap);
            a.q[1] = *(const float4*)(ap + 16);
            const _Float16* bp = &W1t[(size_t)ncol * 128 + kk + boff];
            b.q[0] = *(const float4*)(bp);
            b.q[1] = *(const float4*)(bp + 16);
            acc = __builtin_amdgcn_wmma_f32_16x16x32_f16(
                false, a.v, false, b.v, (short)0, acc, false, false);
        }
        const float bias = b1[ncol];
        #pragma unroll
        for (int r = 0; r < 8; ++r) {
            float vv = acc[r] + bias;
            vv = vv > 0.0f ? vv : 0.0f;
            sHid[(mrow + r) * 256 + ncol] = (_Float16)vv;
        }
    }
    __syncthreads();

    // ---- GEMM2: O = H @ W2 + b2 -> sOut (f32)
    #pragma unroll
    for (int t = 0; t < 2; ++t) {
        const int ncol = (wave * 2 + t) * 16 + (lane & 15);
        v8f acc = {};
        #pragma unroll
        for (int kk = 0; kk < 256; kk += 32) {
            F16x16 a, b;
            const _Float16* ap = &sHid[(lane & 15) * 256 + kk + aoff];
            a.q[0] = *(const float4*)(ap);
            a.q[1] = *(const float4*)(ap + 16);
            const _Float16* bp = &W2t[(size_t)ncol * 256 + kk + boff];
            b.q[0] = *(const float4*)(bp);
            b.q[1] = *(const float4*)(bp + 16);
            acc = __builtin_amdgcn_wmma_f32_16x16x32_f16(
                false, a.v, false, b.v, (short)0, acc, false, false);
        }
        const float bias = b2[ncol];
        #pragma unroll
        for (int r = 0; r < 8; ++r)
            sOut[(mrow + r) * 256 + ncol] = acc[r] + bias;
    }
    __syncthreads();

    // ---- Coalesced writeback (+ optional elementwise add of `memory`)
    for (int v = tid; v < (16 * 256) / 4; v += 256) {
        int e  = v * 4;
        int r  = e >> 8;
        int cc = e & 255;
        size_t g = (size_t)(row0 + r) * 256 + cc;
        float4 o = *(const float4*)&sOut[e];
        if (ADD_MEM) {
            float4 mv = *(const float4*)&mem[g];
            o.x += mv.x; o.y += mv.y; o.z += mv.z; o.w += mv.w;
        }
        *(float4*)&out[g] = o;
    }
}

// ---------------------------------------------------------------------------
// Top-16 nearest neighbors. mem_coor for one batch (8192 pts) cached in LDS.
// One wave per query; per-lane sorted top-16 of u64 keys (d2 bits << 32 | idx)
// -> monotone in d2, ties broken by lower index (matches jax top_k order).
// Merge via 16 shuffle-reduction extraction rounds (sorted ascending output).
// ---------------------------------------------------------------------------
#define TOPK_LMAX 8192

__global__ __launch_bounds__(256)
void topk_kernel(const float* __restrict__ q_coor,    // (B,S,2)
                 const float* __restrict__ mem_coor,  // (B,L,2)
                 int* __restrict__ idx,               // (B,S,16)
                 int S, int L)
{
    __shared__ float sx[TOPK_LMAX];
    __shared__ float sy[TOPK_LMAX];

    const int b  = blockIdx.y;
    const int q0 = blockIdx.x * 64;         // 64 queries per block
    const float* mc = mem_coor + (size_t)b * L * 2;
    for (int i = threadIdx.x; i < L; i += 256) {
        float2 c = ((const float2*)mc)[i];
        sx[i] = c.x; sy[i] = c.y;
    }
    __syncthreads();

    const int lane = threadIdx.x & 31;
    const int wave = threadIdx.x >> 5;

    for (int qi = 0; qi < 8; ++qi) {
        const int s = q0 + wave * 8 + qi;
        const float qx = q_coor[((size_t)b * S + s) * 2 + 0];
        const float qy = q_coor[((size_t)b * S + s) * 2 + 1];

        unsigned long long top[16];
        #pragma unroll
        for (int j = 0; j < 16; ++j) top[j] = 0xFFFFFFFFFFFFFFFFull;

        for (int i = lane; i < L; i += 32) {
            float dx = qx - sx[i];
            float dy = qy - sy[i];
            float d2 = dx * dx + dy * dy;        // >= 0 -> bits monotone
            unsigned long long key =
                ((unsigned long long)__float_as_uint(d2) << 32) | (unsigned)i;
            if (key < top[15]) {                 // sorted ascending, [15]=max
                top[15] = key;
                #pragma unroll
                for (int j = 15; j > 0; --j) {
                    unsigned long long lo = top[j - 1] < top[j] ? top[j - 1] : top[j];
                    unsigned long long hi = top[j - 1] < top[j] ? top[j] : top[j - 1];
                    top[j - 1] = lo; top[j] = hi;
                }
            }
        }

        // merge 32 sorted per-lane lists: 16 rounds of wave-wide min extraction
        int head = 0;
        unsigned long long cur = top[0];
        int outv = 0;
        #pragma unroll 1
        for (int t = 0; t < 16; ++t) {
            unsigned long long m = cur;
            #pragma unroll
            for (int off = 16; off > 0; off >>= 1) {
                unsigned long long o = __shfl_xor(m, off, 32);
                m = o < m ? o : m;
            }
            if (lane == t) outv = (int)(m & 0xFFFFFFFFull);
            if (cur == m) {                      // keys globally unique
                head++;
                unsigned long long nv = 0xFFFFFFFFFFFFFFFFull;
                #pragma unroll
                for (int j = 1; j < 16; ++j)
                    if (j == head) nv = top[j];
                cur = nv;
            }
        }
        if (lane < 16)
            idx[((size_t)b * S + s) * 16 + lane] = outv;
    }
}

// ---------------------------------------------------------------------------
// Gather + softmax(over D) + elementwise output. One wave per query row,
// 8 floats/lane (D=256), loops over k=16 reusing the query registers.
// Pure streaming kernel -> HBM roofline (~1 GB total traffic).
// ---------------------------------------------------------------------------
__global__ __launch_bounds__(256)
void attn_out_kernel(const float* __restrict__ query,   // (B*S, 256)
                     const float* __restrict__ kvpe,    // (B*L, 256)
                     const float* __restrict__ memory,  // (B*L, 256)
                     const int* __restrict__ idx,       // (B*S, 16)
                     float* __restrict__ out,           // (B*S, 16, 256)
                     int S, int L)
{
    const int lane = threadIdx.x & 31;
    const int wave = threadIdx.x >> 5;
    const size_t sq = (size_t)blockIdx.x * 8 + wave;    // global (b*S + s)
    const int b = (int)(sq / (size_t)S);

    const float* qrow = query + sq * 256 + lane * 8;
    const float4 q0 = *(const float4*)(qrow);
    const float4 q1 = *(const float4*)(qrow + 4);
    const float scale = 0.0625f;                        // 1/sqrt(256)

    for (int k = 0; k < 16; ++k) {
        const int id = idx[sq * 16 + k];
        const size_t mr = ((size_t)b * L + id) * 256 + lane * 8;
        const float4 kv0 = *(const float4*)(kvpe + mr);
        const float4 kv1 = *(const float4*)(kvpe + mr + 4);

        float p[8] = { q0.x * kv0.x, q0.y * kv0.y, q0.z * kv0.z, q0.w * kv0.w,
                       q1.x * kv1.x, q1.y * kv1.y, q1.z * kv1.z, q1.w * kv1.w };
        float mx = -3.4e38f;
        #pragma unroll
        for (int r = 0; r < 8; ++r) { p[r] *= scale; mx = fmaxf(mx, p[r]); }
        #pragma unroll
        for (int off = 16; off > 0; off >>= 1)
            mx = fmaxf(mx, __shfl_xor(mx, off, 32));

        float e[8], sum = 0.0f;
        #pragma unroll
        for (int r = 0; r < 8; ++r) { e[r] = __expf(p[r] - mx); sum += e[r]; }
        #pragma unroll
        for (int off = 16; off > 0; off >>= 1)
            sum += __shfl_xor(sum, off, 32);
        const float inv = 1.0f / sum;

        const float4 m0 = *(const float4*)(memory + mr);
        const float4 m1 = *(const float4*)(memory + mr + 4);
        float4 o0, o1;
        o0.x = e[0] * inv * m0.x; o0.y = e[1] * inv * m0.y;
        o0.z = e[2] * inv * m0.z; o0.w = e[3] * inv * m0.w;
        o1.x = e[4] * inv * m1.x; o1.y = e[5] * inv * m1.y;
        o1.z = e[6] * inv * m1.z; o1.w = e[7] * inv * m1.w;

        float* orow = out + (sq * 16 + (size_t)k) * 256 + lane * 8;
        *(float4*)(orow)     = o0;
        *(float4*)(orow + 4) = o1;
    }
}

// ---------------------------------------------------------------------------
// Launch
// ---------------------------------------------------------------------------
extern "C" void kernel_launch(void* const* d_in, const int* in_sizes, int n_in,
                              void* d_out, int out_size, void* d_ws, size_t ws_size,
                              hipStream_t stream)
{
    const float* memory   = (const float*)d_in[0];
    const float* mem_coor = (const float*)d_in[1];
    const float* q_coor   = (const float*)d_in[2];
    const float* Wq1 = (const float*)d_in[3];
    const float* bq1 = (const float*)d_in[4];
    const float* Wq2 = (const float*)d_in[5];
    const float* bq2 = (const float*)d_in[6];
    const float* Wk1 = (const float*)d_in[7];
    const float* bk1 = (const float*)d_in[8];
    const float* Wk2 = (const float*)d_in[9];
    const float* bk2 = (const float*)d_in[10];

    const int B = 4;
    const int D = 256;
    const int S = in_sizes[2] / (B * 2);    // 4096
    const int L = in_sizes[0] / (B * D);    // 8192

    // workspace layout
    char* ws = (char*)d_ws;
    _Float16* Wq1t = (_Float16*)ws;             // 256*128
    _Float16* Wq2t = Wq1t + 256 * 128;          // 256*256
    _Float16* Wk1t = Wq2t + 256 * 256;
    _Float16* Wk2t = Wk1t + 256 * 128;
    float* queryWS = (float*)(Wk2t + 256 * 256);            // (B*S, 256)
    float* kvpeWS  = queryWS + (size_t)B * S * D;           // (B*L, 256)
    int*   idxWS   = (int*)(kvpeWS + (size_t)B * L * D);    // (B*S, 16)

    // 1) weight f16 transpose/convert (L2-resident afterwards)
    convert_wt_kernel<<<(128 * 256 + 255) / 256, 256, 0, stream>>>(Wq1, Wq1t, 128, 256);
    convert_wt_kernel<<<(256 * 256 + 255) / 256, 256, 0, stream>>>(Wq2, Wq2t, 256, 256);
    convert_wt_kernel<<<(128 * 256 + 255) / 256, 256, 0, stream>>>(Wk1, Wk1t, 128, 256);
    convert_wt_kernel<<<(256 * 256 + 255) / 256, 256, 0, stream>>>(Wk2, Wk2t, 256, 256);

    // 2) fused posemb + MLP (WMMA) for queries and for memory (+= memory)
    mlp_pe_kernel<false><<<(B * S) / 16, 256, 0, stream>>>(
        q_coor, Wq1t, bq1, Wq2t, bq2, nullptr, queryWS, B * S);
    mlp_pe_kernel<true><<<(B * L) / 16, 256, 0, stream>>>(
        mem_coor, Wk1t, bk1, Wk2t, bk2, memory, kvpeWS, B * L);

    // 3) top-16 neighbor indices
    topk_kernel<<<dim3(S / 64, B), 256, 0, stream>>>(q_coor, mem_coor, idxWS, S, L);

    // 4) gather + softmax + output
    attn_out_kernel<<<(B * S) / 8, 256, 0, stream>>>(
        queryWS, kvpeWS, memory, idxWS, (float*)d_out, S, L);
}
